// SpMiddleFHD_62010737819736
// MI455X (gfx1250) — matrix operand order repaired and checked by reading, not compiled, not run on hardware
//
#include <hip/hip_runtime.h>

// ---------- CDNA5 types ----------
typedef __attribute__((ext_vector_type(16))) __bf16 v16bf;
typedef __attribute__((ext_vector_type(8)))  __bf16 v8bf;
typedef __attribute__((ext_vector_type(8)))  float  v8f;

union AV { v16bf v; v8bf h[2]; unsigned short u[16]; };

#define BN_SCALE_C 0.99950037f   // 1/sqrt(1.001)
#define SEG 8192
#define NSEGMAX 8

// ---------------------------------------------------------------------------
// Utility fills
// ---------------------------------------------------------------------------
__global__ void fill_u32(unsigned* p, unsigned v, long n) {
    long i = (long)blockIdx.x * blockDim.x + threadIdx.x;
    long stride = (long)gridDim.x * blockDim.x;
    for (; i < n; i += stride) p[i] = v;
}

// ---------------------------------------------------------------------------
// Scatter voxel features -> dense bf16 grid (C stored = 32, real 4), mask0,
// and points_mean into d_out.
// ---------------------------------------------------------------------------
__global__ void scatter_kernel(const float* __restrict__ vf, const int* __restrict__ coors,
                               __bf16* __restrict__ dense, float* __restrict__ mask0,
                               float* __restrict__ pm, int n) {
    int i = blockIdx.x * blockDim.x + threadIdx.x;
    if (i >= n) return;
    int b = coors[i*4+0], z = coors[i*4+1], y = coors[i*4+2], x = coors[i*4+3];
    long vox = ((long)z * 256 + y) * 256 + x;
    long base = vox * 32;
    for (int c = 0; c < 4; ++c) dense[base + c] = (__bf16)vf[i*4 + c];
    mask0[vox] = 1.0f;
    pm[i*4+0] = (float)b;
    pm[i*4+1] = vf[i*4+0];
    pm[i*4+2] = vf[i*4+1];
    pm[i*4+3] = vf[i*4+2];
}

// ---------------------------------------------------------------------------
// Weight prep: f32 [tap][cin_real][cout_real] -> bf16 B-fragment-major
// frag[(((tap*KCH+kc)*CT+ct)*32+lane)*16+j], B[k][n] with
//   k = (j&7) + (lane>=16 ? 8 : 0) + (j>=8 ? 16 : 0), n = ct*16 + (lane&15); zero padded.
// ---------------------------------------------------------------------------
__global__ void prep_weights(const float* __restrict__ w, __bf16* __restrict__ frag,
                             int ntap, int KCH, int CT, int cin_real, int cout_real) {
    int i = blockIdx.x * blockDim.x + threadIdx.x;
    int total = ntap * KCH * CT * 32 * 16;
    if (i >= total) return;
    int j    = i & 15;
    int lane = (i >> 4) & 31;
    int r    = i >> 9;
    int ct   = r % CT; r /= CT;
    int kc   = r % KCH;
    int tap  = r / KCH;
    int k  = (j & 7) + ((lane >= 16) ? 8 : 0) + ((j & 8) ? 16 : 0);
    int ci = kc * 32 + k;
    int co = ct * 16 + (lane & 15);
    float v = (ci < cin_real && co < cout_real)
                  ? w[((long)tap * cin_real + ci) * cout_real + co] : 0.0f;
    frag[i] = (__bf16)v;
}

// ---------------------------------------------------------------------------
// Implicit-GEMM 3D conv, one wave = 16 output x-positions x 16 out channels.
// bf16 in/out, f32 accumulate via v_wmma_f32_16x16x32_bf16. Fused BN*relu*mask.
// Tile geometry is compile-time: KCH = CSin/32, CT = CSout/16, kernel+pad.
// grid = (WT*CT, Hout, Dout)
// ---------------------------------------------------------------------------
template <int KCH, int CT, int KD, int KH, int KW, int PAD>
__global__ void __launch_bounds__(32) conv_wmma(
    const __bf16* __restrict__ in, __bf16* __restrict__ out,
    const __bf16* __restrict__ wfrag, const float* __restrict__ gamma,
    const float* __restrict__ mask,
    int Din, int Hin, int Win,
    int Hout, int Wout, int cout_real,
    int sd, int shw) {
    constexpr int CSin  = KCH * 32;
    constexpr int CSout = CT * 16;
    int lane = threadIdx.x;
    int ct = (int)blockIdx.x & (CT - 1);
    int wt = (int)blockIdx.x / CT;           // CT is power-of-two -> shift
    int y = blockIdx.y;
    int z = blockIdx.z;
    int mrow  = lane & 15;
    int khalf = lane >> 4;
    int xo = wt * 16 + mrow;
    v8f acc = {};
    for (int kd = 0; kd < KD; ++kd) {
        int zin = z * sd + kd - PAD;
        if (zin < 0 || zin >= Din) continue;
        for (int kh = 0; kh < KH; ++kh) {
            int yin = y * shw + kh - PAD;
            if (yin < 0 || yin >= Hin) continue;
            long rowbase = ((long)zin * Hin + yin) * Win;
#pragma unroll
            for (int kw = 0; kw < KW; ++kw) {
                int xin = xo * shw + kw - PAD;
                bool valid = (xo < Wout) && (xin >= 0) && (xin < Win);
                int tap = (kd * KH + kh) * KW + kw;
                long abase = (rowbase + xin) * CSin + 8 * khalf;
#pragma unroll
                for (int kc = 0; kc < KCH; ++kc) {
                    AV a;
                    if (valid) {
                        const __bf16* p = in + abase + kc * 32;
                        a.h[0] = *(const v8bf*)(p);
                        a.h[1] = *(const v8bf*)(p + 16);
                    } else {
#pragma unroll
                        for (int j = 0; j < 16; ++j) a.u[j] = 0;
                    }
                    v16bf b = *(const v16bf*)(wfrag + (((long)(tap * KCH + kc) * CT + ct) * 32 + lane) * 16);
                    acc = __builtin_amdgcn_wmma_f32_16x16x32_bf16(
                        false, a.v, false, b, (short)0, acc, false, false);
                }
            }
        }
    }
    int co = ct * 16 + (lane & 15);
    float g = (co < cout_real) ? gamma[co] * BN_SCALE_C : 0.0f;
#pragma unroll
    for (int i = 0; i < 8; ++i) {
        int m2 = i + 8 * khalf;
        int xo2 = wt * 16 + m2;
        if (xo2 >= Wout) continue;
        float v = acc[i] * g;
        v = v > 0.0f ? v : 0.0f;
        long oidx = ((long)z * Hout + y) * Wout + xo2;
        if (mask) v *= mask[oidx];
        out[oidx * CSout + co] = (__bf16)v;
    }
}

// ---------------------------------------------------------------------------
// Mask max-pool 3x3x3 stride 2 pad 1
// ---------------------------------------------------------------------------
__global__ void down_mask_k(const float* __restrict__ mi, float* __restrict__ mo,
                            int Di, int Hi, int Wi, int Do, int Ho, int Wo) {
    int i = blockIdx.x * blockDim.x + threadIdx.x;
    int n = Do * Ho * Wo;
    if (i >= n) return;
    int x = i % Wo, y = (i / Wo) % Ho, z = i / (Wo * Ho);
    float m = 0.0f;
    for (int dz = 0; dz < 3; ++dz) {
        int zi = z * 2 + dz - 1; if (zi < 0 || zi >= Di) continue;
        for (int dy = 0; dy < 3; ++dy) {
            int yi = y * 2 + dy - 1; if (yi < 0 || yi >= Hi) continue;
            for (int dx = 0; dx < 3; ++dx) {
                int xi = x * 2 + dx - 1; if (xi < 0 || xi >= Wi) continue;
                m = fmaxf(m, mi[((long)zi * Hi + yi) * Wi + xi]);
            }
        }
    }
    mo[i] = m;
}

// ---------------------------------------------------------------------------
// Compact occupied voxels -> (kxyz, kfeat). Unfilled kxyz slots stay at 1e9.
// ---------------------------------------------------------------------------
__global__ void known_gather(const float* __restrict__ mask, const __bf16* __restrict__ feat,
                             int CS, int creal, int Dl, int Hl, int Wl,
                             float vsx, float vsy, float vsz, int K,
                             float* __restrict__ kxyz, float* __restrict__ kfeat,
                             int* __restrict__ counter) {
    int i = blockIdx.x * blockDim.x + threadIdx.x;
    int n = Dl * Hl * Wl;
    if (i >= n) return;
    if (mask[i] <= 0.0f) return;
    int x = i % Wl, y = (i / Wl) % Hl, z = i / (Wl * Hl);
    int idx = atomicAdd(counter, 1);
    if (idx >= K) return;
    kxyz[idx*3+0] = x * vsx + 0.0f  + 0.5f * vsx;
    kxyz[idx*3+1] = y * vsy - 40.0f + 0.5f * vsy;
    kxyz[idx*3+2] = z * vsz - 3.0f  + 0.5f * vsz;
    for (int c = 0; c < creal; ++c)
        kfeat[(long)idx * creal + c] = (float)feat[(long)i * CS + c];
}

// ---------------------------------------------------------------------------
// Segmented brute-force top-3 NN: LDS-tiled known points, per (point, segment)
// ---------------------------------------------------------------------------
__global__ void topk_partial(const float* __restrict__ vf, int npts,
                             const float* __restrict__ kxyz, int K, int seg_size,
                             float* __restrict__ pd, int* __restrict__ pi, int nsegmax) {
    __shared__ float sx[256], sy[256], sz[256];
    int s = blockIdx.y;
    int pid = blockIdx.x * blockDim.x + threadIdx.x;
    float px = 0.f, py = 0.f, pz = 0.f;
    if (pid < npts) { px = vf[pid*4+0]; py = vf[pid*4+1]; pz = vf[pid*4+2]; }
    float d0 = 3e38f, d1 = 3e38f, d2 = 3e38f;
    int   i0 = 0, i1 = 0, i2 = 0;
    int kbeg = s * seg_size;
    int kend = min(K, kbeg + seg_size);
    for (int t = kbeg; t < kend; t += 256) {
        int j = t + threadIdx.x;
        if (j < kend) { sx[threadIdx.x] = kxyz[j*3+0]; sy[threadIdx.x] = kxyz[j*3+1]; sz[threadIdx.x] = kxyz[j*3+2]; }
        if (t + 256 < kend) __builtin_prefetch(kxyz + (long)(t + 256) * 3, 0, 1);
        __syncthreads();
        int lim = min(256, kend - t);
        for (int u = 0; u < lim; ++u) {
            float dx = px - sx[u], dy = py - sy[u], dz = pz - sz[u];
            float d = dx*dx + dy*dy + dz*dz;
            int gi = t + u;
            if (d < d2) {
                if (d < d1) {
                    d2 = d1; i2 = i1;
                    if (d < d0) { d1 = d0; i1 = i0; d0 = d; i0 = gi; }
                    else        { d1 = d;  i1 = gi; }
                } else { d2 = d; i2 = gi; }
            }
        }
        __syncthreads();
    }
    if (pid < npts) {
        int b = (pid * nsegmax + s) * 3;
        pd[b+0] = d0; pd[b+1] = d1; pd[b+2] = d2;
        pi[b+0] = i0; pi[b+1] = i1; pi[b+2] = i2;
    }
}

// ---------------------------------------------------------------------------
// Merge segment winners, IDW-weight 3 nearest, gather features -> P (bf16)
// blockDim = 64 (one block per point; each thread handles one channel)
// ---------------------------------------------------------------------------
__global__ void interp_merge(const float* __restrict__ pd, const int* __restrict__ pi,
                             int nseg, int nsegmax, const float* __restrict__ kfeat,
                             int creal, __bf16* __restrict__ P, int pofs, int npts) {
    int pid = blockIdx.x;
    if (pid >= npts) return;
    int c = threadIdx.x;
    float d0 = 3e38f, d1 = 3e38f, d2 = 3e38f;
    int   i0 = 0, i1 = 0, i2 = 0;
    for (int s = 0; s < nseg; ++s) {
        int b = (pid * nsegmax + s) * 3;
        for (int t = 0; t < 3; ++t) {
            float d = pd[b + t];
            int   gi = pi[b + t];
            if (d < d2) {
                if (d < d1) {
                    d2 = d1; i2 = i1;
                    if (d < d0) { d1 = d0; i1 = i0; d0 = d; i0 = gi; }
                    else        { d1 = d;  i1 = gi; }
                } else { d2 = d; i2 = gi; }
            }
        }
    }
    float r0 = 1.0f / (d0 + 1e-8f);
    float r1 = 1.0f / (d1 + 1e-8f);
    float r2 = 1.0f / (d2 + 1e-8f);
    float rs = r0 + r1 + r2;
    if (c < creal) {
        float v = kfeat[(long)i0 * creal + c] * (r0 / rs)
                + kfeat[(long)i1 * creal + c] * (r1 / rs)
                + kfeat[(long)i2 * creal + c] * (r2 / rs);
        P[(long)pid * 160 + pofs + c] = (__bf16)v;
    }
}

// ---------------------------------------------------------------------------
// P[M x Kstored] (bf16, row-major) @ fragment-major W -> out f32 [M x 64]
// grid = (Mtiles, CT)
// ---------------------------------------------------------------------------
__global__ void __launch_bounds__(32) gemm_wmma(
    const __bf16* __restrict__ A, const __bf16* __restrict__ bfrag,
    float* __restrict__ out, int npts, int KCH, int CT, int Kstored) {
    int lane = threadIdx.x;
    int mt = blockIdx.x;
    int ct = blockIdx.y;
    int m = mt * 16 + (lane & 15);
    int khalf = lane >> 4;
    v8f acc = {};
    for (int kc = 0; kc < KCH; ++kc) {
        AV a;
        if (m < npts) {
            const __bf16* p = A + (long)m * Kstored + kc * 32 + 8 * khalf;
            a.h[0] = *(const v8bf*)p;
            a.h[1] = *(const v8bf*)(p + 16);
        } else {
            for (int j = 0; j < 16; ++j) a.u[j] = 0;
        }
        v16bf b = *(const v16bf*)(bfrag + ((long)(kc * CT + ct) * 32 + lane) * 16);
        acc = __builtin_amdgcn_wmma_f32_16x16x32_bf16(
            false, a.v, false, b, (short)0, acc, false, false);
    }
    int co = ct * 16 + (lane & 15);
#pragma unroll
    for (int i = 0; i < 8; ++i) {
        int row = mt * 16 + i + 8 * khalf;
        if (row < npts) out[(long)row * 64 + co] = acc[i];
    }
}

// ---------------------------------------------------------------------------
// Heads: cls (64->1) and reg (64->3)
// ---------------------------------------------------------------------------
__global__ void heads_k(const float* __restrict__ pw, const float* __restrict__ wcls,
                        const float* __restrict__ wreg, float* __restrict__ ocls,
                        float* __restrict__ oreg, int n) {
    int i = blockIdx.x * blockDim.x + threadIdx.x;
    if (i >= n) return;
    float c = 0.f, r0 = 0.f, r1 = 0.f, r2 = 0.f;
    for (int k = 0; k < 64; ++k) {
        float v = pw[(long)i * 64 + k];
        c  += v * wcls[k];
        r0 += v * wreg[k*3+0];
        r1 += v * wreg[k*3+1];
        r2 += v * wreg[k*3+2];
    }
    ocls[i] = c;
    oreg[i*3+0] = r0; oreg[i*3+1] = r1; oreg[i*3+2] = r2;
}

// ---------------------------------------------------------------------------
// Final transpose/reshape: h (2,32,32,64) bf16 -> ret (128,32,32) f32
// ---------------------------------------------------------------------------
__global__ void ret_k(const __bf16* __restrict__ h, float* __restrict__ ret) {
    int i = blockIdx.x * blockDim.x + threadIdx.x;
    if (i >= 131072) return;
    int x  = i & 31;
    int y  = (i >> 5) & 31;
    int zz = (i >> 10) & 1;
    int co = i >> 11;
    ret[i] = (float)h[(((long)(zz * 32 + y)) * 32 + x) * 64 + co];
}

// ---------------------------------------------------------------------------
// Host-side conv dispatch (template instantiation selection)
// ---------------------------------------------------------------------------
static void launch_conv(hipStream_t stream, __bf16* wfragp,
                        const float* w, const float* g, const float* mask,
                        const __bf16* in, __bf16* outb,
                        int Din, int Hin, int Win, int CSin, int cin_real,
                        int Dout, int Hout, int Wout, int CSout, int cout_real,
                        int KD, int KH, int KW, int pad, int sd, int shw) {
    int ntap = KD * KH * KW;
    int KCH = CSin / 32;
    int CT = CSout / 16;
    int prep_n = ntap * KCH * CT * 512;
    prep_weights<<<(prep_n + 255) / 256, 256, 0, stream>>>(
        w, wfragp, ntap, KCH, CT, cin_real, cout_real);
    int WT = (Wout + 15) / 16;
    dim3 grid((unsigned)(WT * CT), (unsigned)Hout, (unsigned)Dout);
    if (KD == 3) {
        if (KCH == 1 && CT == 2)
            conv_wmma<1, 2, 3, 3, 3, 1><<<grid, 32, 0, stream>>>(
                in, outb, wfragp, g, mask, Din, Hin, Win, Hout, Wout, cout_real, sd, shw);
        else if (KCH == 1 && CT == 4)
            conv_wmma<1, 4, 3, 3, 3, 1><<<grid, 32, 0, stream>>>(
                in, outb, wfragp, g, mask, Din, Hin, Win, Hout, Wout, cout_real, sd, shw);
        else
            conv_wmma<2, 4, 3, 3, 3, 1><<<grid, 32, 0, stream>>>(
                in, outb, wfragp, g, mask, Din, Hin, Win, Hout, Wout, cout_real, sd, shw);
    } else {
        conv_wmma<2, 4, 2, 1, 1, 0><<<grid, 32, 0, stream>>>(
            in, outb, wfragp, g, mask, Din, Hin, Win, Hout, Wout, cout_real, sd, shw);
    }
}

static unsigned f2u_host(float f) { union { float f; unsigned u; } c; c.f = f; return c.u; }

extern "C" void kernel_launch(void* const* d_in, const int* in_sizes, int n_in,
                              void* d_out, int out_size, void* d_ws, size_t ws_size,
                              hipStream_t stream) {
    const float* vf    = (const float*)d_in[0];
    const int*   coors = (const int*)d_in[1];
    const float* w0a = (const float*)d_in[3];  const float* g0a = (const float*)d_in[4];
    const float* w0b = (const float*)d_in[5];  const float* g0b = (const float*)d_in[6];
    const float* wd0 = (const float*)d_in[7];  const float* gd0 = (const float*)d_in[8];
    const float* w1a = (const float*)d_in[9];  const float* g1a = (const float*)d_in[10];
    const float* w1b = (const float*)d_in[11]; const float* g1b = (const float*)d_in[12];
    const float* wd1 = (const float*)d_in[13]; const float* gd1 = (const float*)d_in[14];
    const float* w2a = (const float*)d_in[15]; const float* g2a = (const float*)d_in[16];
    const float* w2b = (const float*)d_in[17]; const float* g2b = (const float*)d_in[18];
    const float* w2c = (const float*)d_in[19]; const float* g2c = (const float*)d_in[20];
    const float* wd2 = (const float*)d_in[21]; const float* gd2 = (const float*)d_in[22];
    const float* w3a = (const float*)d_in[23]; const float* g3a = (const float*)d_in[24];
    const float* w3b = (const float*)d_in[25]; const float* g3b = (const float*)d_in[26];
    const float* w3c = (const float*)d_in[27]; const float* g3c = (const float*)d_in[28];
    const float* wen = (const float*)d_in[29]; const float* gen = (const float*)d_in[30];
    const float* wfc = (const float*)d_in[31];
    const float* wcls = (const float*)d_in[32];
    const float* wreg = (const float*)d_in[33];

    int npts = in_sizes[0] / 4;

    // ---- workspace carve ----
    size_t off = 0;
    auto carve = [&](size_t bytes) -> char* {
        char* p = (char*)d_ws + off;
        off += (bytes + 255) & ~(size_t)255;
        return p;
    };
    __bf16* bufA  = (__bf16*)carve(167772160);      // 40*256*256*32 bf16
    __bf16* bufB  = (__bf16*)carve(167772160);
    float* mask0  = (float*)carve(10485760);        // 40*256*256 f32
    float* mask1  = (float*)carve(1310720);         // 20*128*128
    float* mask2  = (float*)carve(163840);          // 10*64*64
    float* mask3  = (float*)carve(20480);           // 5*32*32
    __bf16* wfragp = (__bf16*)carve(262144);
    float* kxyz   = (float*)carve(64000L * 3 * 4);
    float* kfeat  = (float*)carve(64000L * 64 * 4);
    int*   counter = (int*)carve(256);
    float* pd     = (float*)carve((size_t)npts * NSEGMAX * 3 * 4);
    int*   pi     = (int*)carve((size_t)npts * NSEGMAX * 3 * 4);
    __bf16* P     = (__bf16*)carve((size_t)npts * 160 * 2);
    float* pwbuf  = (float*)carve((size_t)npts * 64 * 4);

    float* outf    = (float*)d_out;
    float* out_ret = outf;              // 131072
    float* out_pm  = outf + 131072;     // 32000
    float* out_cls = outf + 163072;     // 8000
    float* out_reg = outf + 171072;     // 24000

    // ---- known + interp helper ----
    auto run_known_interp = [&](const float* maskl, const __bf16* feat, int CS, int creal,
                                int Dl, int Hl, int Wl, int level, int K, int pofs) {
        float s = (float)(1 << level);
        float vsx = 0.05f * s, vsy = 0.05f * s, vsz = 0.1f * s;
        fill_u32<<<256, 256, 0, stream>>>((unsigned*)kxyz, f2u_host(1e9f), (long)K * 3);
        fill_u32<<<1, 32, 0, stream>>>((unsigned*)counter, 0u, 1);
        int nv = Dl * Hl * Wl;
        known_gather<<<(nv + 255) / 256, 256, 0, stream>>>(
            maskl, feat, CS, creal, Dl, Hl, Wl, vsx, vsy, vsz, K, kxyz, kfeat, counter);
        int nseg = (K + SEG - 1) / SEG;
        dim3 g((unsigned)((npts + 255) / 256), (unsigned)nseg);
        topk_partial<<<g, 256, 0, stream>>>(vf, npts, kxyz, K, SEG, pd, pi, NSEGMAX);
        interp_merge<<<npts, 64, 0, stream>>>(pd, pi, nseg, NSEGMAX, kfeat, creal, P, pofs, npts);
    };

    // ---- pipeline ----
    fill_u32<<<4096, 256, 0, stream>>>((unsigned*)bufA, 0u, 41943040L);   // dense zeros
    fill_u32<<<4096, 256, 0, stream>>>((unsigned*)mask0, 0u, 2621440L);
    scatter_kernel<<<(npts + 255) / 256, 256, 0, stream>>>(vf, coors, bufA, mask0, out_pm, npts);

    // level 0
    launch_conv(stream, wfragp, w0a, g0a, mask0, bufA, bufB, 40,256,256,32,4,  40,256,256,32,16, 3,3,3,1,1,1);
    launch_conv(stream, wfragp, w0b, g0b, mask0, bufB, bufA, 40,256,256,32,16, 40,256,256,32,16, 3,3,3,1,1,1);
    launch_conv(stream, wfragp, wd0, gd0, 0,     bufA, bufB, 40,256,256,32,16, 20,128,128,32,32, 3,3,3,1,2,2);
    down_mask_k<<<(20*128*128 + 255)/256, 256, 0, stream>>>(mask0, mask1, 40,256,256, 20,128,128);

    // level 1
    launch_conv(stream, wfragp, w1a, g1a, mask1, bufB, bufA, 20,128,128,32,32, 20,128,128,32,32, 3,3,3,1,1,1);
    launch_conv(stream, wfragp, w1b, g1b, mask1, bufA, bufB, 20,128,128,32,32, 20,128,128,32,32, 3,3,3,1,1,1);
    run_known_interp(mask1, bufB, 32, 32, 20,128,128, 1, 64000, 0);
    launch_conv(stream, wfragp, wd1, gd1, 0,     bufB, bufA, 20,128,128,32,32, 10,64,64,64,64,   3,3,3,1,2,2);
    down_mask_k<<<(10*64*64 + 255)/256, 256, 0, stream>>>(mask1, mask2, 20,128,128, 10,64,64);

    // level 2
    launch_conv(stream, wfragp, w2a, g2a, mask2, bufA, bufB, 10,64,64,64,64, 10,64,64,64,64, 3,3,3,1,1,1);
    launch_conv(stream, wfragp, w2b, g2b, mask2, bufB, bufA, 10,64,64,64,64, 10,64,64,64,64, 3,3,3,1,1,1);
    launch_conv(stream, wfragp, w2c, g2c, mask2, bufA, bufB, 10,64,64,64,64, 10,64,64,64,64, 3,3,3,1,1,1);
    run_known_interp(mask2, bufB, 64, 64, 10,64,64, 2, 40960, 32);
    launch_conv(stream, wfragp, wd2, gd2, 0,     bufB, bufA, 10,64,64,64,64, 5,32,32,64,64,  3,3,3,1,2,2);
    down_mask_k<<<(5*32*32 + 255)/256, 256, 0, stream>>>(mask2, mask3, 10,64,64, 5,32,32);

    // level 3
    launch_conv(stream, wfragp, w3a, g3a, mask3, bufA, bufB, 5,32,32,64,64, 5,32,32,64,64, 3,3,3,1,1,1);
    launch_conv(stream, wfragp, w3b, g3b, mask3, bufB, bufA, 5,32,32,64,64, 5,32,32,64,64, 3,3,3,1,1,1);
    launch_conv(stream, wfragp, w3c, g3c, mask3, bufA, bufB, 5,32,32,64,64, 5,32,32,64,64, 3,3,3,1,1,1);
    run_known_interp(mask3, bufB, 64, 64, 5,32,32, 3, 5120, 96);

    // extra conv (2,1,1) stride (2,1,1), no pad -> (2,32,32,64)
    launch_conv(stream, wfragp, wen, gen, 0, bufB, bufA, 5,32,32,64,64, 2,32,32,64,64, 2,1,1,0,2,1);
    ret_k<<<512, 256, 0, stream>>>(bufA, out_ret);

    // pw = P(8000x160) @ wfc(160x64) using bf16 WMMA
    prep_weights<<<(1 * 5 * 4 * 512 + 255) / 256, 256, 0, stream>>>(wfc, wfragp, 1, 5, 4, 160, 64);
    dim3 ggrid((unsigned)((npts + 15) / 16), 4u);
    gemm_wmma<<<ggrid, 32, 0, stream>>>(P, wfragp, pwbuf, npts, 5, 4, 160);

    // heads
    heads_k<<<(npts + 255) / 256, 256, 0, stream>>>(pwbuf, wcls, wreg, out_cls, out_reg, npts);
}
